// HybridEpisodicMemory_4793183502805
// MI455X (gfx1250) — matrix operand order repaired and checked by reading, not compiled, not run on hardware
//
#include <hip/hip_runtime.h>
#include <hip/hip_bf16.h>
#include <math.h>

#define BATCHN 1024
#define DIMQ   1024
#define LATD   128
#define HOTN   16384
#define COLDN  262144
#define CHUNK  2048              // keys per workgroup chunk (8 waves x 256 keys)
#define NCH_H  (HOTN / CHUNK)    // 8
#define NCH_C  (COLDN / CHUNK)   // 128
#define MQ     32                // queries per workgroup (2 x 16 M-tiles)

typedef __bf16 bf16_t;
typedef __attribute__((ext_vector_type(16))) __bf16 v16bf;
typedef __attribute__((ext_vector_type(8)))  float  v8f;

// sorted-descending top-4 insertion; strict '>' keeps earliest index on ties
__device__ __forceinline__ void topk_insert(float s, int id,
    float& t0, float& t1, float& t2, float& t3,
    int& i0, int& i1, int& i2, int& i3) {
  if (s > t2) {
    t3 = t2; i3 = i2;
    if (s > t1) {
      t2 = t1; i2 = i1;
      if (s > t0) { t1 = t0; i1 = i0; t0 = s; i0 = id; }
      else        { t1 = s;  i1 = id; }
    } else        { t2 = s;  i2 = id; }
  } else if (s > t3) { t3 = s; i3 = id; }
}

// ---------------- q = l2norm(query); also emit bf16 copy ------------------
__global__ __launch_bounds__(256)
void normalize_q(const float* __restrict__ q, float* __restrict__ qf,
                 bf16_t* __restrict__ qb) {
  __shared__ float red[256];
  const int b = blockIdx.x, tid = threadIdx.x;
  const float* row = q + (size_t)b * DIMQ;
  float v[4]; float ss = 0.f;
#pragma unroll
  for (int i = 0; i < 4; ++i) { v[i] = row[tid + i * 256]; ss += v[i] * v[i]; }
  red[tid] = ss; __syncthreads();
  for (int off = 128; off > 0; off >>= 1) {
    if (tid < off) red[tid] += red[tid + off];
    __syncthreads();
  }
  const float inv = 1.0f / fmaxf(sqrtf(red[0]), 1e-12f);
#pragma unroll
  for (int i = 0; i < 4; ++i) {
    const float x = v[i] * inv;
    qf[(size_t)b * DIMQ + tid + i * 256] = x;
    qb[(size_t)b * DIMQ + tid + i * 256] = (bf16_t)x;
  }
}

// ---------------- q_lat = l2norm(q @ key_proj) ----------------------------
__global__ __launch_bounds__(128)
void qlat_kernel(const float* __restrict__ qf, const float* __restrict__ kp,
                 float* __restrict__ qlf, bf16_t* __restrict__ qlb) {
  __shared__ float lq[DIMQ];
  __shared__ float red[128];
  const int b = blockIdx.x, tid = threadIdx.x;
  for (int i = tid; i < DIMQ; i += 128) lq[i] = qf[(size_t)b * DIMQ + i];
  __syncthreads();
  float acc = 0.f;
  for (int d = 0; d < DIMQ; ++d) acc += lq[d] * kp[(size_t)d * LATD + tid];
  red[tid] = acc * acc; __syncthreads();
  for (int off = 64; off > 0; off >>= 1) {
    if (tid < off) red[tid] += red[tid + off];
    __syncthreads();
  }
  const float inv = 1.0f / fmaxf(sqrtf(red[0]), 1e-12f);
  const float x = acc * inv;
  qlf[(size_t)b * LATD + tid] = x;
  qlb[(size_t)b * LATD + tid] = (bf16_t)x;
}

// ------- similarity (bf16 WMMA, f32 acc) + fused per-chunk top-4 ----------
// M-blocked x2: one B fragment feeds two WMMAs (32 queries per WG).
// grid: (BATCHN/MQ, nkeys/CHUNK), block 256 (8 wave32)
template <int K>
__global__ __launch_bounds__(256)
void sim_topk_chunk(const bf16_t* __restrict__ qbf, const float* __restrict__ keys,
                    int nchunks, float* __restrict__ cs, int* __restrict__ ci) {
  __shared__ float tile[8 * 2 * 256];   // per-wave two 16x16 score tiles
  __shared__ float mscore[8 * 32 * 4];
  __shared__ int   midx[8 * 32 * 4];

  const int tid  = threadIdx.x;
  const int wave = tid >> 5;
  const int lane = tid & 31;
  const int half = lane >> 4;   // 0: lanes 0-15, 1: lanes 16-31
  const int mrow = lane & 15;
  const int qbase = blockIdx.x * MQ;
  const int waveKeyBase = blockIdx.y * CHUNK + wave * 256;

  // A operands (ISA 16-bit A 16x32 layout): two query rows per lane
  const bf16_t* aptr0 = qbf + (size_t)(qbase + mrow) * K;        // M-tile 0
  const bf16_t* aptr1 = aptr0 + (size_t)16 * K;                  // M-tile 1

  float t0 = -1e30f, t1 = -1e30f, t2 = -1e30f, t3 = -1e30f;
  int   i0 = 0, i1 = 0, i2 = 0, i3 = 0;
  float* wtile = tile + wave * 512;

  for (int t = 0; t < 16; ++t) {          // 16 key-tiles of 16 keys per wave
    const int ktile = waveKeyBase + t * 16;
    const float* bptr = keys + (size_t)(ktile + mrow) * K;

    // prefetch next key-tile's rows (256B granularity per lane pair)
    if (t < 15) {
      const float* npt = keys + (size_t)(ktile + 16 + mrow) * K;
#pragma unroll
      for (int j = 0; j < K; j += 128)
        __builtin_prefetch((const void*)(npt + j + half * 64), 0, 1);
    }

    v8f acc0 = {}, acc1 = {};
    for (int kb = 0; kb < K; kb += 32) {
      union { v16bf v; uint4 u[2]; } a0, a1;
      a0.u[0] = *(const uint4*)(aptr0 + kb + half * 8);
      a0.u[1] = *(const uint4*)(aptr0 + kb + 16 + half * 8);
      a1.u[0] = *(const uint4*)(aptr1 + kb + half * 8);
      a1.u[1] = *(const uint4*)(aptr1 + kb + 16 + half * 8);
      const float4* bf4 = (const float4*)(bptr + kb + half * 8);
      const float4* bf4h = (const float4*)(bptr + kb + 16 + half * 8);
      const float4 f0 = bf4[0], f1 = bf4[1], f2 = bf4h[0], f3 = bf4h[1];
      v16bf bv;
      bv[0]  = (bf16_t)f0.x; bv[1]  = (bf16_t)f0.y; bv[2]  = (bf16_t)f0.z; bv[3]  = (bf16_t)f0.w;
      bv[4]  = (bf16_t)f1.x; bv[5]  = (bf16_t)f1.y; bv[6]  = (bf16_t)f1.z; bv[7]  = (bf16_t)f1.w;
      bv[8]  = (bf16_t)f2.x; bv[9]  = (bf16_t)f2.y; bv[10] = (bf16_t)f2.z; bv[11] = (bf16_t)f2.w;
      bv[12] = (bf16_t)f3.x; bv[13] = (bf16_t)f3.y; bv[14] = (bf16_t)f3.z; bv[15] = (bf16_t)f3.w;
      acc0 = __builtin_amdgcn_wmma_f32_16x16x32_bf16(
          false, a0.v, false, bv, (short)0, acc0, false, false);
      acc1 = __builtin_amdgcn_wmma_f32_16x16x32_bf16(
          false, a1.v, false, bv, (short)0, acc1, false, false);
    }

    // Wave-private spill + scan: per-wave LDS ops are in-order (DScnt), so a
    // wavefront-scope fence (compiler ordering) is sufficient -- no WG barrier.
    __builtin_amdgcn_fence(__ATOMIC_ACQ_REL, "wavefront"); // prior scan vs stores
    // D layout: VGPR i -> (M = 8*half + i, N = mrow)
#pragma unroll
    for (int i = 0; i < 8; ++i) {
      wtile[(half * 8 + i) * 16 + mrow]       = acc0[i];
      wtile[256 + (half * 8 + i) * 16 + mrow] = acc1[i];
    }
    __builtin_amdgcn_fence(__ATOMIC_ACQ_REL, "wavefront"); // stores vs scan loads

    // all 32 lanes scan: lane owns query row (half*16 + mrow) of its M-group
    const float* rowp = wtile + half * 256 + mrow * 16;
#pragma unroll
    for (int j = 0; j < 16; ++j)
      topk_insert(rowp[j], ktile + j, t0, t1, t2, t3, i0, i1, i2, i3);
  }

  {
    const int base = (wave * 32 + lane) * 4;
    mscore[base + 0] = t0; midx[base + 0] = i0;
    mscore[base + 1] = t1; midx[base + 1] = i1;
    mscore[base + 2] = t2; midx[base + 2] = i2;
    mscore[base + 3] = t3; midx[base + 3] = i3;
  }
  __syncthreads();

  if (wave == 0) {   // 32 lanes merge the 8 waves' top-4s (one query each)
    float s0 = -1e30f, s1 = -1e30f, s2 = -1e30f, s3 = -1e30f;
    int   j0 = 0, j1 = 0, j2 = 0, j3 = 0;
    for (int w = 0; w < 8; ++w)
      for (int s = 0; s < 4; ++s) {
        const int base = (w * 32 + lane) * 4 + s;
        topk_insert(mscore[base], midx[base], s0, s1, s2, s3, j0, j1, j2, j3);
      }
    const int b = qbase + half * 16 + mrow;
    const int o = (b * nchunks + blockIdx.y) * 4;
    cs[o + 0] = s0; ci[o + 0] = j0;
    cs[o + 1] = s1; ci[o + 1] = j1;
    cs[o + 2] = s2; ci[o + 2] = j2;
    cs[o + 3] = s3; ci[o + 3] = j3;
  }
}

// ---------------- merge per-chunk top-4s into global top-4 ----------------
__global__ __launch_bounds__(256)
void merge_chunks(const float* __restrict__ cs, const int* __restrict__ ci,
                  int nchunks, float* __restrict__ S, int* __restrict__ I) {
  const int b = blockIdx.x * 256 + threadIdx.x;
  if (b >= BATCHN) return;
  float t0 = -1e30f, t1 = -1e30f, t2 = -1e30f, t3 = -1e30f;
  int   i0 = 0, i1 = 0, i2 = 0, i3 = 0;
  for (int c = 0; c < nchunks; ++c)
    for (int s = 0; s < 4; ++s) {
      const int o = (b * nchunks + c) * 4 + s;
      topk_insert(cs[o], ci[o], t0, t1, t2, t3, i0, i1, i2, i3);
    }
  S[b * 4 + 0] = t0; I[b * 4 + 0] = i0;
  S[b * 4 + 1] = t1; I[b * 4 + 1] = i1;
  S[b * 4 + 2] = t2; I[b * 4 + 2] = i2;
  S[b * 4 + 3] = t3; I[b * 4 + 3] = i3;
}

// ---------------- hot gather + hot scores ---------------------------------
__global__ __launch_bounds__(256)
void gather_hot(const float* __restrict__ hk, const float* __restrict__ hv,
                const float* __restrict__ S, const int* __restrict__ I,
                float* __restrict__ outK, float* __restrict__ outV,
                float* __restrict__ outS) {
  const int b = blockIdx.x, slot = blockIdx.y;
  const int idx = I[b * 4 + slot];
  const float* kr = hk + (size_t)idx * DIMQ;
  const float* vr = hv + (size_t)idx * DIMQ;
  float* ok = outK + ((size_t)b * 8 + slot) * DIMQ;
  float* ov = outV + ((size_t)b * 8 + slot) * DIMQ;
  for (int d = threadIdx.x; d < DIMQ; d += 256) { ok[d] = kr[d]; ov[d] = vr[d]; }
  if (threadIdx.x == 0) outS[b * 8 + slot] = S[b * 4 + slot];
}

// ------- cold gather + back-projection + l2norm + fallback mask -----------
__global__ __launch_bounds__(256)
void cold_project(const float* __restrict__ ck, const float* __restrict__ cv,
                  const float* __restrict__ kp, const float* __restrict__ vp,
                  const float* __restrict__ Sh, const float* __restrict__ Sc,
                  const int* __restrict__ Ic,
                  float* __restrict__ outK, float* __restrict__ outV,
                  float* __restrict__ outS) {
  __shared__ float kcs[LATD], vcs[LATD];
  __shared__ float red[256];
  const int b = blockIdx.x, slot = blockIdx.y, tid = threadIdx.x;
  const bool fb = Sh[b * 4 + 0] < 0.35f;   // S sorted desc -> [0] is the max
  const int idx = Ic[b * 4 + slot];
  if (tid < LATD) {
    kcs[tid] = ck[(size_t)idx * LATD + tid];
    vcs[tid] = cv[(size_t)idx * LATD + tid];
  }
  __syncthreads();

  float accK[4], accV[4]; float ss = 0.f;
#pragma unroll
  for (int i = 0; i < 4; ++i) {
    const int d = tid + i * 256;
    const float* kpr = kp + (size_t)d * LATD;
    const float* vpr = vp + (size_t)d * LATD;
    float ak = 0.f, av = 0.f;
    for (int l = 0; l < LATD; ++l) { ak += kcs[l] * kpr[l]; av += vcs[l] * vpr[l]; }
    accK[i] = ak; accV[i] = av; ss += ak * ak;
  }
  red[tid] = ss; __syncthreads();
  for (int off = 128; off > 0; off >>= 1) {
    if (tid < off) red[tid] += red[tid + off];
    __syncthreads();
  }
  const float nrm = sqrtf(red[0]);
  const float sk = fb ? (1.0f / fmaxf(nrm, 1e-12f)) : 0.0f;
  const float sv = fb ? 1.0f : 0.0f;
  float* ok = outK + ((size_t)b * 8 + 4 + slot) * DIMQ;
  float* ov = outV + ((size_t)b * 8 + 4 + slot) * DIMQ;
#pragma unroll
  for (int i = 0; i < 4; ++i) {
    const int d = tid + i * 256;
    ok[d] = accK[i] * sk;
    ov[d] = accV[i] * sv;
  }
  if (tid == 0) outS[b * 8 + 4 + slot] = fb ? Sc[b * 4 + slot] : 0.0f;
}

extern "C" void kernel_launch(void* const* d_in, const int* in_sizes, int n_in,
                              void* d_out, int out_size, void* d_ws, size_t ws_size,
                              hipStream_t stream) {
  const float* query       = (const float*)d_in[0];
  const float* hot_keys    = (const float*)d_in[1];
  const float* hot_values  = (const float*)d_in[2];
  const float* cold_keys   = (const float*)d_in[3];
  const float* cold_values = (const float*)d_in[4];
  const float* key_proj    = (const float*)d_in[5];
  const float* value_proj  = (const float*)d_in[6];

  float* out  = (float*)d_out;
  float* outK = out;
  float* outV = out + (size_t)BATCHN * 8 * DIMQ;
  float* outS = out + (size_t)2 * BATCHN * 8 * DIMQ;

  char* w = (char*)d_ws;
  auto take = [&](size_t bytes) -> char* {
    char* p = w; w += (bytes + 255) & ~(size_t)255; return p;
  };
  float*  qf    = (float*)take((size_t)BATCHN * DIMQ * 4);
  bf16_t* qb    = (bf16_t*)take((size_t)BATCHN * DIMQ * 2);
  float*  qlf   = (float*)take((size_t)BATCHN * LATD * 4);
  bf16_t* qlb   = (bf16_t*)take((size_t)BATCHN * LATD * 2);
  float*  csH   = (float*)take((size_t)BATCHN * NCH_H * 4 * 4);
  int*    ciH   = (int*)take((size_t)BATCHN * NCH_H * 4 * 4);
  float*  csC   = (float*)take((size_t)BATCHN * NCH_C * 4 * 4);
  int*    ciC   = (int*)take((size_t)BATCHN * NCH_C * 4 * 4);
  float*  Sh    = (float*)take((size_t)BATCHN * 4 * 4);
  int*    Ih    = (int*)take((size_t)BATCHN * 4 * 4);
  float*  Scold = (float*)take((size_t)BATCHN * 4 * 4);
  int*    Icold = (int*)take((size_t)BATCHN * 4 * 4);

  normalize_q<<<BATCHN, 256, 0, stream>>>(query, qf, qb);
  qlat_kernel<<<BATCHN, 128, 0, stream>>>(qf, key_proj, qlf, qlb);

  sim_topk_chunk<DIMQ><<<dim3(BATCHN / MQ, NCH_H), 256, 0, stream>>>(
      qb, hot_keys, NCH_H, csH, ciH);
  sim_topk_chunk<LATD><<<dim3(BATCHN / MQ, NCH_C), 256, 0, stream>>>(
      qlb, cold_keys, NCH_C, csC, ciC);

  merge_chunks<<<BATCHN / 256, 256, 0, stream>>>(csH, ciH, NCH_H, Sh, Ih);
  merge_chunks<<<BATCHN / 256, 256, 0, stream>>>(csC, ciC, NCH_C, Scold, Icold);

  gather_hot<<<dim3(BATCHN, 4), 256, 0, stream>>>(hot_keys, hot_values, Sh, Ih,
                                                  outK, outV, outS);
  cold_project<<<dim3(BATCHN, 4), 256, 0, stream>>>(cold_keys, cold_values,
                                                    key_proj, value_proj,
                                                    Sh, Scold, Icold,
                                                    outK, outV, outS);
}
